// AutoShape_68247030334476
// MI455X (gfx1250) — compile-verified, matrix operand order checked by reading, not run on hardware
//
#include <hip/hip_runtime.h>
#include <hip/hip_bf16.h>

typedef __attribute__((ext_vector_type(16))) __bf16 v16bf;
typedef __attribute__((ext_vector_type(8)))  float  v8f;

#define NCCLS   80
#define NOUT    85
#define OTOT    255
#define TOTROWS 25200
#define NBATCH  16
#define CONF    0.25f
#define IOUTH   0.45f
#define MAXDET  300
#define MAXWH   4096.0f

struct AnchS { float a[6]; };   // (w,h) per anchor, already multiplied by stride

// ---------------------------------------------------------------------------
// Pack fp32 weights (255 x C) into bf16 WMMA B-fragment order:
// flat index = ((kt*16 + nt)*32 + lane)*16 + i
// lane<16 holds K {0..7, 16..23}; lane>=16 holds K {8..15, 24..31}  (ISA 7.12.2)
// ---------------------------------------------------------------------------
__global__ void pack_w_kernel(const float* __restrict__ w, __bf16* __restrict__ wp,
                              int C, int total) {
    int t = blockIdx.x * blockDim.x + threadIdx.x;
    if (t >= total) return;
    int i    = t & 15;
    int lane = (t >> 4) & 31;
    int nt   = (t >> 9) & 15;
    int kt   = t >> 13;
    int o = nt * 16 + (lane & 15);
    int k = kt * 32 + ((lane < 16) ? 0 : 8) + ((i < 8) ? i : (8 + i));
    float v = (o < OTOT) ? w[(size_t)o * C + k] : 0.0f;
    wp[t] = (__bf16)v;
}

// ---------------------------------------------------------------------------
// Fused 1x1 conv (bf16 WMMA, fp32 accum) + bias + sigmoid + YOLO decode.
// Block: 512 threads = 16 waves; wave nt owns output tile [nt*16, nt*16+16).
// Block covers 64 pixels (4 M-tiles of 16) of image blockIdx.y.
// Double-buffered LDS A-tile AND software-pipelined B fragment: the WMMAs of
// step kt consume only values fetched during step kt-1, so the per-iteration
// wait never drains the freshly issued next-step loads.
// ---------------------------------------------------------------------------
__global__ __launch_bounds__(512)
void conv_decode_kernel(const float* __restrict__ x, const __bf16* __restrict__ wp,
                        const float* __restrict__ bias, float* __restrict__ pred,
                        int C, int nx, int P, int lvl_off, float stride, AnchS anch)
{
    __shared__ v16bf afrag[2][4][32];       // double-buffered, fragment-ordered, 8 KB

    const int tid  = threadIdx.x;
    const int lane = tid & 31;
    const int nt   = tid >> 5;              // wave id == N-tile
    const int b    = blockIdx.y;
    const int p0   = blockIdx.x * 64;

    // Loader role: thread loads 4 consecutive pixels of one channel (float4).
    const int c  = tid >> 4;                // channel within K-step, 0..31
    const int m4 = (tid & 15) * 4;          // first pixel within 64-pixel tile
    int pld = p0 + m4; if (pld + 3 >= P) pld = P - 4;   // P is a multiple of 4
    const float* xc = x + (size_t)b * C * P + (size_t)c * P + pld;

    // LDS scatter targets: fragment layout (ISA 7.12.2); pixel -> lane stride 16 bf16
    const int mt = m4 >> 4;
    const int dl = (m4 & 15) + ((c & 8) ? 16 : 0);
    const int di = (c & 7) + ((c & 16) ? 8 : 0);
    __bf16* dst0 = ((__bf16*)&afrag[0][mt][0]) + dl * 16 + di;
    __bf16* dst1 = ((__bf16*)&afrag[1][mt][0]) + dl * 16 + di;

    const int NK = C >> 5;
    v8f acc[4] = {};

    // Prologue: stage K-step 0 into buffer 0, preload B fragment for step 0
    {
        float4 v4 = *(const float4*)xc;
        dst0[ 0] = (__bf16)v4.x;
        dst0[16] = (__bf16)v4.y;
        dst0[32] = (__bf16)v4.z;
        dst0[48] = (__bf16)v4.w;
    }
    v16bf bfrag_cur = *(const v16bf*)(wp + ((size_t)nt * 32 + lane) * 16);
    __syncthreads();

    for (int kt = 0; kt < NK; ++kt) {
        const bool has_next = (kt + 1) < NK;

        // 1) Issue next K-step's loads early (A tile + B fragment); they are
        //    consumed only after the next barrier, hiding latency under WMMA.
        float4 nv;
        v16bf bfrag_next = bfrag_cur;
        if (has_next) {
            nv = *(const float4*)(xc + (size_t)(kt + 1) * 32 * P);
            bfrag_next = *(const v16bf*)(wp + (((size_t)(kt + 1) * 16 + nt) * 32 + lane) * 16);
        }
        if (kt + 2 < NK)                    // keep a prefetch two steps ahead
            __builtin_prefetch(xc + (size_t)(kt + 2) * 32 * P, 0, 1);

        // 2) Pull all 4 A fragments (8x ds_load_b128) before the WMMA chain
        const v16bf* abuf = &afrag[kt & 1][0][0];
        v16bf af0 = abuf[0 * 32 + lane];
        v16bf af1 = abuf[1 * 32 + lane];
        v16bf af2 = abuf[2 * 32 + lane];
        v16bf af3 = abuf[3 * 32 + lane];

        // 3) 4 WMMAs, operands all pre-fetched
        acc[0] = __builtin_amdgcn_wmma_f32_16x16x32_bf16(
                     false, af0, false, bfrag_cur, (short)0, acc[0], false, false);
        acc[1] = __builtin_amdgcn_wmma_f32_16x16x32_bf16(
                     false, af1, false, bfrag_cur, (short)0, acc[1], false, false);
        acc[2] = __builtin_amdgcn_wmma_f32_16x16x32_bf16(
                     false, af2, false, bfrag_cur, (short)0, acc[2], false, false);
        acc[3] = __builtin_amdgcn_wmma_f32_16x16x32_bf16(
                     false, af3, false, bfrag_cur, (short)0, acc[3], false, false);

        // 4) Convert + scatter next A tile into the other buffer
        if (has_next) {
            __bf16* d = ((kt + 1) & 1) ? dst1 : dst0;
            d[ 0] = (__bf16)nv.x;
            d[16] = (__bf16)nv.y;
            d[32] = (__bf16)nv.z;
            d[48] = (__bf16)nv.w;
        }
        bfrag_cur = bfrag_next;
        __syncthreads();                    // single barrier per K-step
    }

    // Epilogue: D layout -> N = lane%16, M = j + 8*(lane>=16)
    int o = nt * 16 + (lane & 15);
    if (o < OTOT) {
        float bv = bias[o];
        int ai = o / NOUT;
        int oo = o - ai * NOUT;
        float aw = anch.a[ai * 2 + 0];
        float ah = anch.a[ai * 2 + 1];
        int mhi = (lane >= 16) ? 8 : 0;
        #pragma unroll
        for (int t = 0; t < 4; ++t) {
            #pragma unroll
            for (int j = 0; j < 8; ++j) {
                int p = p0 + t * 16 + j + mhi;
                if (p < P) {
                    float v = acc[t][j] + bv;
                    float s = 1.0f / (1.0f + __expf(-v));   // v_exp_f32
                    float outv;
                    if      (oo == 0) outv = (2.0f * s + (float)(p % nx) - 0.5f) * stride;
                    else if (oo == 1) outv = (2.0f * s + (float)(p / nx) - 0.5f) * stride;
                    else if (oo == 2) { float q = 2.0f * s; outv = q * q * aw; }
                    else if (oo == 3) { float q = 2.0f * s; outv = q * q * ah; }
                    else              outv = s;
                    pred[((size_t)b * TOTROWS + lvl_off + (size_t)ai * P + p) * NOUT + oo] = outv;
                }
            }
        }
    }
}

// ---------------------------------------------------------------------------
// Per-image NMS: gather > CONF candidates, bitonic sort by score (desc),
// greedy class-offset IoU suppression, write top-300 dets.
// ---------------------------------------------------------------------------
__global__ __launch_bounds__(1024)
void nms_kernel(const float* __restrict__ pred, float* __restrict__ dets)
{
    __shared__ float  s_score[1024];
    __shared__ float4 s_box[1024];
    __shared__ float  s_cls[1024];
    __shared__ int    s_keep[1024];
    __shared__ int    s_cnt;

    const int tid = threadIdx.x;
    const int b   = blockIdx.x;
    if (tid == 0) s_cnt = 0;
    s_score[tid] = 0.0f;
    s_box[tid]   = make_float4(0.f, 0.f, 0.f, 0.f);
    s_cls[tid]   = 0.0f;
    __syncthreads();

    const float* pb = pred + (size_t)b * TOTROWS * NOUT;
    for (int i = tid; i < TOTROWS; i += 1024) {
        const float* r = pb + (size_t)i * NOUT;
        float obj = r[4];
        float mc = 0.0f; int mcls = 0;
        for (int cc = 0; cc < NCCLS; ++cc) {
            float v = r[5 + cc];
            if (v > mc) { mc = v; mcls = cc; }
        }
        float sc = obj * mc;
        if (sc > CONF) {
            int slot = atomicAdd(&s_cnt, 1);
            if (slot < 1024) {
                float xcx = r[0], ycy = r[1], w = r[2], h = r[3];
                s_score[slot] = sc;
                s_box[slot]   = make_float4(xcx - 0.5f * w, ycy - 0.5f * h,
                                            xcx + 0.5f * w, ycy + 0.5f * h);
                s_cls[slot]   = (float)mcls;
            }
        }
    }
    __syncthreads();

    // bitonic sort, descending by score
    for (int k = 2; k <= 1024; k <<= 1) {
        for (int j = k >> 1; j > 0; j >>= 1) {
            int ixj = tid ^ j;
            if (ixj > tid) {
                bool up = ((tid & k) == 0);
                float s1 = s_score[tid], s2 = s_score[ixj];
                if (up ? (s1 < s2) : (s1 > s2)) {
                    s_score[tid] = s2; s_score[ixj] = s1;
                    float4 t4 = s_box[tid]; s_box[tid] = s_box[ixj]; s_box[ixj] = t4;
                    float tc = s_cls[tid]; s_cls[tid] = s_cls[ixj]; s_cls[ixj] = tc;
                }
            }
            __syncthreads();
        }
    }

    s_keep[tid] = (s_score[tid] > CONF) ? 1 : 0;
    __syncthreads();

    // greedy suppression (serial over rank, parallel over victims)
    for (int i = 0; i < 1024; ++i) {
        if (s_keep[i] && tid > i && s_keep[tid]) {
            float oi = s_cls[i]   * MAXWH;
            float ot = s_cls[tid] * MAXWH;
            float4 bi = s_box[i], bt = s_box[tid];
            float ax1 = bi.x + oi, ay1 = bi.y + oi, ax2 = bi.z + oi, ay2 = bi.w + oi;
            float bx1 = bt.x + ot, by1 = bt.y + ot, bx2 = bt.z + ot, by2 = bt.w + ot;
            float lx = fmaxf(ax1, bx1), ly = fmaxf(ay1, by1);
            float rx = fminf(ax2, bx2), ry = fminf(ay2, by2);
            float iw = fmaxf(rx - lx, 0.0f), ih = fmaxf(ry - ly, 0.0f);
            float inter = iw * ih;
            float areaA = (ax2 - ax1) * (ay2 - ay1);
            float areaB = (bx2 - bx1) * (by2 - by1);
            float iou = inter / (areaA + areaB - inter + 1e-7f);
            if (iou > IOUTH) s_keep[tid] = 0;
        }
        __syncthreads();
    }

    if (tid == 0) {
        float* db = dets + (size_t)b * MAXDET * 6;
        int m = 0;
        for (int i = 0; i < 1024 && m < MAXDET; ++i) {
            if (s_keep[i]) {
                float4 bx = s_box[i];
                db[m * 6 + 0] = bx.x; db[m * 6 + 1] = bx.y;
                db[m * 6 + 2] = bx.z; db[m * 6 + 3] = bx.w;
                db[m * 6 + 4] = s_score[i]; db[m * 6 + 5] = s_cls[i];
                ++m;
            }
        }
        for (; m < MAXDET; ++m)
            for (int q = 0; q < 6; ++q) db[m * 6 + q] = 0.0f;
    }
}

// ---------------------------------------------------------------------------
extern "C" void kernel_launch(void* const* d_in, const int* in_sizes, int n_in,
                              void* d_out, int out_size, void* d_ws, size_t ws_size,
                              hipStream_t stream) {
    const float* x0 = (const float*)d_in[0];
    const float* x1 = (const float*)d_in[1];
    const float* x2 = (const float*)d_in[2];
    const float* w0 = (const float*)d_in[3];
    const float* b0 = (const float*)d_in[4];
    const float* w1 = (const float*)d_in[5];
    const float* b1 = (const float*)d_in[6];
    const float* w2 = (const float*)d_in[7];
    const float* b2 = (const float*)d_in[8];

    float* dets = (float*)d_out;
    float* pred = (float*)d_out + (size_t)NBATCH * MAXDET * 6;

    // packed bf16 weights in workspace: (C/32)*8192 elems per level
    __bf16* wp0 = (__bf16*)d_ws;          //  32768 elems (C=128)
    __bf16* wp1 = wp0 + 32768;            //  65536 elems (C=256)
    __bf16* wp2 = wp0 + 98304;            // 131072 elems (C=512)

    pack_w_kernel<<<(32768  + 255) / 256, 256, 0, stream>>>(w0, wp0, 128, 32768);
    pack_w_kernel<<<(65536  + 255) / 256, 256, 0, stream>>>(w1, wp1, 256, 65536);
    pack_w_kernel<<<(131072 + 255) / 256, 256, 0, stream>>>(w2, wp2, 512, 131072);

    AnchS a0 = {{ 10.f*8.f, 13.f*8.f, 16.f*8.f, 30.f*8.f, 33.f*8.f, 23.f*8.f }};
    AnchS a1 = {{ 30.f*16.f, 61.f*16.f, 62.f*16.f, 45.f*16.f, 59.f*16.f, 119.f*16.f }};
    AnchS a2 = {{ 116.f*32.f, 90.f*32.f, 156.f*32.f, 198.f*32.f, 373.f*32.f, 326.f*32.f }};

    conv_decode_kernel<<<dim3((6400 + 63) / 64, NBATCH), 512, 0, stream>>>(
        x0, wp0, b0, pred, 128, 80, 6400, 0,     8.0f,  a0);
    conv_decode_kernel<<<dim3((1600 + 63) / 64, NBATCH), 512, 0, stream>>>(
        x1, wp1, b1, pred, 256, 40, 1600, 19200, 16.0f, a1);
    conv_decode_kernel<<<dim3(( 400 + 63) / 64, NBATCH), 512, 0, stream>>>(
        x2, wp2, b2, pred, 512, 20,  400, 24000, 32.0f, a2);

    nms_kernel<<<NBATCH, 1024, 0, stream>>>(pred, dets);
}
